// SelfAttention_18227841204710
// MI455X (gfx1250) — matrix operand order repaired and checked by reading, not compiled, not run on hardware
//
#include <hip/hip_runtime.h>
#include <hip/hip_bf16.h>

typedef __attribute__((ext_vector_type(16))) __bf16 v16bf;
typedef __attribute__((ext_vector_type(8)))  float  v8f;
typedef unsigned short ushort_t;
typedef unsigned int   uint_t;
typedef int v4i_t __attribute__((vector_size(16)));   // matches builtin's V4i param

#define WMMA_BF16(A, B, C) \
  __builtin_amdgcn_wmma_f32_16x16x32_bf16(false, (A), false, (B), (short)0, (C), false, false)

// ---------- async LDS staging (gfx1250 GLOBAL_LOAD_ASYNC_TO_LDS, ASYNCcnt) ----------
#if defined(__has_builtin)
#  if __has_builtin(__builtin_amdgcn_global_load_async_to_lds_b128)
#    define HAVE_ASYNC_LDS 1
#  endif
#endif
#ifndef HAVE_ASYNC_LDS
#  define HAVE_ASYNC_LDS 0
#endif

__device__ __forceinline__ void async_copy16(void* lds_dst, const void* gsrc) {
#if HAVE_ASYNC_LDS
  __builtin_amdgcn_global_load_async_to_lds_b128(
      (__attribute__((address_space(1))) v4i_t*)gsrc,
      (__attribute__((address_space(3))) v4i_t*)lds_dst, 0, 0);
#else
  *(uint4*)lds_dst = *(const uint4*)gsrc;          // synchronous fallback
#endif
}
__device__ __forceinline__ void async_wait_all() {
#if HAVE_ASYNC_LDS
  asm volatile("s_wait_asynccnt 0x0" ::: "memory");
#endif
}

// ---------- scalar helpers ----------
__device__ __forceinline__ ushort_t f2bf(float x) {
  uint_t u = __float_as_uint(x);
  uint_t r = u + 0x7FFFu + ((u >> 16) & 1u);       // round-to-nearest-even
  return (ushort_t)(r >> 16);
}
__device__ __forceinline__ uint_t f2bf2(float lo, float hi) {
  return (uint_t)f2bf(lo) | ((uint_t)f2bf(hi) << 16);
}

// A-fragment: 16x32 bf16, lane holds row=lane&15, K per ISA table.
__device__ __forceinline__ v16bf load_af(const ushort_t* base, int stride, int lane, int koff) {
  const int row = lane & 15, half = lane >> 4;
  union { uint_t u[8]; v16bf v; } f;
  const ushort_t* r = base + (size_t)row * stride;
#pragma unroll
  for (int j = 0; j < 8; ++j) {
    int k = koff + ((j < 4) ? (half * 8 + 2 * j) : (16 + half * 8 + 2 * (j - 4)));
    f.u[j] = *(const uint_t*)(r + k);
  }
  return f.v;
}

// B-fragment: 32x16 bf16 (KxN), lane holds col n=lane&15, K per ISA table (same as A).
__device__ __forceinline__ v16bf load_bf(const ushort_t* base, int stride, int lane, int koff) {
  const int row = lane & 15, half = lane >> 4;
  union { uint_t u[8]; v16bf v; } f;
  const ushort_t* r = base + (size_t)row * stride;
#pragma unroll
  for (int j = 0; j < 8; ++j) {
    int k = koff + ((j < 4) ? (half * 8 + 2 * j) : (16 + half * 8 + 2 * (j - 4)));
    f.u[j] = *(const uint_t*)(r + k);
  }
  return f.v;
}

// ---------- sizes ----------
#define BATCH 2
#define SEQ   2048
#define CDIM  768
#define HEADS 12
#define HDIM  64
#define MTOT  (BATCH * SEQ)          // 4096

// ================= Kernel 0: f32 -> bf16 conversion pre-pass =================
__global__ __launch_bounds__(256) void cvt_bf16(const float* __restrict__ s,
                                                ushort_t* __restrict__ d, int n4) {
  int i = blockIdx.x * 256 + threadIdx.x;
  if (i < n4) {
    float4 x = ((const float4*)s)[i];
    uint2 o; o.x = f2bf2(x.x, x.y); o.y = f2bf2(x.z, x.w);
    ((uint2*)d)[i] = o;
  }
}

// ================= Kernel 1: QKV projection =================
// QKV[m,d] = sum_c Xbf[m,c]*Wabf[d,c]. Q,K -> [B,H,W,64] bf16 (Q*=1/8); V -> [B,H,64,W] (transposed).
__global__ __launch_bounds__(256) void qkv_gemm(
    const ushort_t* __restrict__ Abf, const ushort_t* __restrict__ Bbf,
    ushort_t* __restrict__ Qb, ushort_t* __restrict__ Kb, ushort_t* __restrict__ VTb) {
  constexpr int BM = 128, BN = 64, BK = 64;
  __shared__ ushort_t sA[2][BM * BK];   // [m][k]
  __shared__ ushort_t sB[2][BN * BK];   // [n][k]

  const int tid = threadIdx.x, lane = tid & 31, wid = tid >> 5;
  const int wm = wid & 3, wn = wid >> 2;
  const int m0 = blockIdx.y * BM, n0 = blockIdx.x * BN;

  auto stage = [&](int buf, int kc) {
    for (int p = tid; p < BM * BK / 8; p += 256) {          // 16B chunks
      int row = p >> 3, k = (p & 7) << 3;
      async_copy16(&sA[buf][row * BK + k], &Abf[(size_t)(m0 + row) * CDIM + kc + k]);
    }
    for (int p = tid; p < BN * BK / 8; p += 256) {
      int row = p >> 3, k = (p & 7) << 3;
      async_copy16(&sB[buf][row * BK + k], &Bbf[(size_t)(n0 + row) * CDIM + kc + k]);
    }
  };

  v8f acc[2][2] = {};
  stage(0, 0);
  int it = 0;
  for (int kc = 0; kc < CDIM; kc += BK, ++it) {
    const int cur = it & 1;
    async_wait_all();
    __syncthreads();                        // tile `cur` ready; prev compute done everywhere
    if (kc + BK < CDIM) stage(cur ^ 1, kc + BK);
#pragma unroll
    for (int ks = 0; ks < 2; ++ks) {
      v16bf a0 = load_af(&sA[cur][(wm * 32) * BK],      BK, lane, ks * 32);
      v16bf a1 = load_af(&sA[cur][(wm * 32 + 16) * BK], BK, lane, ks * 32);
      v16bf b0 = load_bf(&sB[cur][(wn * 32) * BK],      BK, lane, ks * 32);
      v16bf b1 = load_bf(&sB[cur][(wn * 32 + 16) * BK], BK, lane, ks * 32);
      acc[0][0] = WMMA_BF16(a0, b0, acc[0][0]);
      acc[0][1] = WMMA_BF16(a0, b1, acc[0][1]);
      acc[1][0] = WMMA_BF16(a1, b0, acc[1][0]);
      acc[1][1] = WMMA_BF16(a1, b1, acc[1][1]);
    }
  }

  const int col = lane & 15, half = lane >> 4;
#pragma unroll
  for (int i = 0; i < 2; ++i)
#pragma unroll
    for (int j = 0; j < 2; ++j) {
      int d = n0 + wn * 32 + j * 16 + col;
      int which = d / CDIM, cc = d % CDIM;
      int h = cc >> 6, kd = cc & 63;
      float s = (which == 0) ? 0.125f : 1.0f;   // 1/sqrt(64) folded into Q
#pragma unroll
      for (int r = 0; r < 8; ++r) {
        int m = m0 + wm * 32 + i * 16 + r + 8 * half;
        int b = m >> 11, w = m & 2047;
        size_t bh = (size_t)(b * HEADS + h);
        ushort_t v = f2bf(acc[i][j][r] * s);
        if (which == 0)      Qb [(bh * SEQ + w) * HDIM + kd] = v;
        else if (which == 1) Kb [(bh * SEQ + w) * HDIM + kd] = v;
        else                 VTb[(bh * HDIM + kd) * SEQ + w] = v;   // transposed
      }
    }
}

// ================= Kernel 2: causal flash attention =================
__global__ __launch_bounds__(256) void attn_kernel(
    const ushort_t* __restrict__ Q, const ushort_t* __restrict__ K,
    const ushort_t* __restrict__ VT, ushort_t* __restrict__ Out) {
  __shared__ ushort_t sK[2][32 * 64];    // [key][kd]
  __shared__ ushort_t sVT[2][64 * 32];   // [kd][key]
  __shared__ ushort_t sP[8][16 * 32];    // per-wave P tile [qrow][keycol]

  const int tid = threadIdx.x, lane = tid & 31, wid = tid >> 5;
  const int col = lane & 15, half = lane >> 4;
  const int qblk = blockIdx.x & 15, bh = blockIdx.x >> 4;   // bh = b*12 + h
  const int b = bh / HEADS, h = bh % HEADS;
  const int qb0 = qblk * 128, q0 = qb0 + wid * 16;

  const ushort_t* Qh  = Q  + (size_t)bh * SEQ * HDIM;
  const ushort_t* Kh  = K  + (size_t)bh * SEQ * HDIM;
  const ushort_t* VTh = VT + (size_t)bh * HDIM * SEQ;

  v16bf aq0 = load_af(Qh + (size_t)q0 * HDIM, HDIM, lane, 0);
  v16bf aq1 = load_af(Qh + (size_t)q0 * HDIM, HDIM, lane, 32);

  auto stageKV = [&](int buf, int kb) {
    for (int p = tid; p < 32 * 64 / 8; p += 256) {          // K rows, 16B chunks
      int row = p >> 3, k = (p & 7) << 3;
      async_copy16(&sK[buf][row * 64 + k], &Kh[(size_t)(kb + row) * HDIM + k]);
    }
    for (int p = tid; p < 64 * 32 / 8; p += 256) {          // V^T rows, 16B chunks
      int kd = p >> 2, kk = (p & 3) << 3;
      async_copy16(&sVT[buf][kd * 32 + kk], &VTh[(size_t)kd * SEQ + kb + kk]);
    }
  };

  float mrow[8], lrow[8];
#pragma unroll
  for (int r = 0; r < 8; ++r) { mrow[r] = -3.0e38f; lrow[r] = 0.0f; }
  v8f o[4] = {};

  const int nkb = qb0 + 128;
  stageKV(0, 0);
  int it = 0;
  for (int kb = 0; kb < nkb; kb += 32, ++it) {
    const int cur = it & 1;
    async_wait_all();
    __syncthreads();
    if (kb + 32 < nkb) stageKV(cur ^ 1, kb + 32);

    if (kb <= q0 + 15) {
      v8f c0 = {}, c1 = {};
      c0 = WMMA_BF16(aq0, load_bf(&sK[cur][0],       64, lane, 0),  c0);
      c0 = WMMA_BF16(aq1, load_bf(&sK[cur][0],       64, lane, 32), c0);
      c1 = WMMA_BF16(aq0, load_bf(&sK[cur][16 * 64], 64, lane, 0),  c1);
      c1 = WMMA_BF16(aq1, load_bf(&sK[cur][16 * 64], 64, lane, 32), c1);

      if (kb + 31 > q0) {
#pragma unroll
        for (int r = 0; r < 8; ++r) {
          int qr = q0 + r + 8 * half;
          if (kb + col > qr)      c0[r] = -3.0e38f;
          if (kb + 16 + col > qr) c1[r] = -3.0e38f;
        }
      }

#pragma unroll
      for (int r = 0; r < 8; ++r) {
        float rm = fmaxf(c0[r], c1[r]);
        rm = fmaxf(rm, __shfl_xor(rm, 1, 32));
        rm = fmaxf(rm, __shfl_xor(rm, 2, 32));
        rm = fmaxf(rm, __shfl_xor(rm, 4, 32));
        rm = fmaxf(rm, __shfl_xor(rm, 8, 32));
        float mnew = fmaxf(mrow[r], rm);
        float sc = __expf(mrow[r] - mnew);
        float p0 = __expf(c0[r] - mnew);
        float p1 = __expf(c1[r] - mnew);
        float rs = p0 + p1;
        rs += __shfl_xor(rs, 1, 32);
        rs += __shfl_xor(rs, 2, 32);
        rs += __shfl_xor(rs, 4, 32);
        rs += __shfl_xor(rs, 8, 32);
        lrow[r] = lrow[r] * sc + rs;
        mrow[r] = mnew;
#pragma unroll
        for (int t = 0; t < 4; ++t) o[t][r] *= sc;
        int prow = r + 8 * half;
        sP[wid][prow * 32 + col]      = f2bf(p0);
        sP[wid][prow * 32 + 16 + col] = f2bf(p1);
      }
      asm volatile("s_wait_dscnt 0x0" ::: "memory");   // same-wave LDS RAW on sP
      v16bf ap = load_af(sP[wid], 32, lane, 0);
#pragma unroll
      for (int t = 0; t < 4; ++t) {
        v16bf bv = load_bf(&sVT[cur][t * 16 * 32], 32, lane, 0);
        o[t] = WMMA_BF16(ap, bv, o[t]);
      }
    }
  }

#pragma unroll
  for (int r = 0; r < 8; ++r) {
    float inv = 1.0f / lrow[r];
    int m = b * SEQ + q0 + r + 8 * half;
#pragma unroll
    for (int t = 0; t < 4; ++t) {
      int c = h * HDIM + t * 16 + col;
      Out[(size_t)m * CDIM + c] = f2bf(o[t][r] * inv);
    }
  }
}

// ================= Kernel 3: output projection =================
__global__ __launch_bounds__(256) void proj_gemm(
    const ushort_t* __restrict__ Abf, const ushort_t* __restrict__ Bbf,
    float* __restrict__ Outp) {
  constexpr int BM = 128, BN = 64, BK = 64;
  __shared__ ushort_t sA[2][BM * BK];
  __shared__ ushort_t sB[2][BN * BK];

  const int tid = threadIdx.x, lane = tid & 31, wid = tid >> 5;
  const int wm = wid & 3, wn = wid >> 2;
  const int m0 = blockIdx.y * BM, n0 = blockIdx.x * BN;

  auto stage = [&](int buf, int kc) {
    for (int p = tid; p < BM * BK / 8; p += 256) {
      int row = p >> 3, k = (p & 7) << 3;
      async_copy16(&sA[buf][row * BK + k], &Abf[(size_t)(m0 + row) * CDIM + kc + k]);
    }
    for (int p = tid; p < BN * BK / 8; p += 256) {
      int row = p >> 3, k = (p & 7) << 3;
      async_copy16(&sB[buf][row * BK + k], &Bbf[(size_t)(n0 + row) * CDIM + kc + k]);
    }
  };

  v8f acc[2][2] = {};
  stage(0, 0);
  int it = 0;
  for (int kc = 0; kc < CDIM; kc += BK, ++it) {
    const int cur = it & 1;
    async_wait_all();
    __syncthreads();
    if (kc + BK < CDIM) stage(cur ^ 1, kc + BK);
#pragma unroll
    for (int ks = 0; ks < 2; ++ks) {
      v16bf a0 = load_af(&sA[cur][(wm * 32) * BK],      BK, lane, ks * 32);
      v16bf a1 = load_af(&sA[cur][(wm * 32 + 16) * BK], BK, lane, ks * 32);
      v16bf b0 = load_bf(&sB[cur][(wn * 32) * BK],      BK, lane, ks * 32);
      v16bf b1 = load_bf(&sB[cur][(wn * 32 + 16) * BK], BK, lane, ks * 32);
      acc[0][0] = WMMA_BF16(a0, b0, acc[0][0]);
      acc[0][1] = WMMA_BF16(a0, b1, acc[0][1]);
      acc[1][0] = WMMA_BF16(a1, b0, acc[1][0]);
      acc[1][1] = WMMA_BF16(a1, b1, acc[1][1]);
    }
  }

  const int col = lane & 15, half = lane >> 4;
#pragma unroll
  for (int i = 0; i < 2; ++i)
#pragma unroll
    for (int j = 0; j < 2; ++j) {
      int d = n0 + wn * 32 + j * 16 + col;
#pragma unroll
      for (int r = 0; r < 8; ++r) {
        int m = m0 + wm * 32 + i * 16 + r + 8 * half;
        Outp[(size_t)m * CDIM + d] = acc[i][j][r];
      }
    }
}

// ================= host launch =================
extern "C" void kernel_launch(void* const* d_in, const int* in_sizes, int n_in,
                              void* d_out, int out_size, void* d_ws, size_t ws_size,
                              hipStream_t stream) {
  const float* X  = (const float*)d_in[0];   // [2,2048,768]
  const float* Wa = (const float*)d_in[1];   // [2304,768]
  const float* Wp = (const float*)d_in[2];   // [768,768]
  float* out = (float*)d_out;                // [2,2048,768]

  const size_t QKV_ELEMS = (size_t)BATCH * HEADS * SEQ * HDIM;  // 3,145,728
  const size_t XC_ELEMS  = (size_t)MTOT * CDIM;                 // 3,145,728
  const size_t WA_ELEMS  = (size_t)3 * CDIM * CDIM;             // 1,769,472

  ushort_t* qb   = (ushort_t*)d_ws;
  ushort_t* kbuf = qb   + QKV_ELEMS;
  ushort_t* vtb  = kbuf + QKV_ELEMS;
  ushort_t* xbf  = vtb  + QKV_ELEMS;   // X bf16, later reused as attention output
  ushort_t* wbuf = xbf  + XC_ELEMS;    // Wa bf16, later reused for Wp bf16 (~28.7MB total)

  const int nX4  = (int)(XC_ELEMS / 4);
  const int nWa4 = (int)(WA_ELEMS / 4);
  const int nWp4 = (int)((size_t)CDIM * CDIM / 4);

  cvt_bf16<<<(nX4  + 255) / 256, 256, 0, stream>>>(X,  xbf,  nX4);
  cvt_bf16<<<(nWa4 + 255) / 256, 256, 0, stream>>>(Wa, wbuf, nWa4);
  qkv_gemm<<<dim3(3 * CDIM / 64, MTOT / 128), 256, 0, stream>>>(xbf, wbuf, qb, kbuf, vtb);
  cvt_bf16<<<(nWp4 + 255) / 256, 256, 0, stream>>>(Wp, wbuf, nWp4);
  attn_kernel<<<BATCH * HEADS * (SEQ / 128), 256, 0, stream>>>(qb, kbuf, vtb, xbf);
  proj_gemm<<<dim3(CDIM / 64, MTOT / 128), 256, 0, stream>>>(xbf, wbuf, out);
}
